// MTREncoder_67860483277518
// MI455X (gfx1250) — matrix operand order repaired and checked by reading, not compiled, not run on hardware
//
#include <hip/hip_runtime.h>
#include <hip/hip_bf16.h>

// ---------------- problem constants (from setup_inputs) ----------------
#define BB   16
#define NN   512
#define DD   256
#define HH   8
#define LL   6
#define KK   16
#define DFF  1024
#define BN   (BB * NN)     // 8192
#define HD   (DD / HH)     // 32

typedef __bf16 bf16x16 __attribute__((ext_vector_type(16)));
typedef __bf16 bf16x8  __attribute__((ext_vector_type(8)));
typedef float  v8f     __attribute__((ext_vector_type(8)));

__device__ __forceinline__ __bf16 f2bf(float f) {
    union { float f; unsigned u; } a; a.f = f;
    unsigned r = (a.u + 0x7FFFu + ((a.u >> 16) & 1u)) >> 16;
    union { unsigned short s; __bf16 b; } o; o.s = (unsigned short)r;
    return o.b;
}

// ---------------- KNN: top-16 nearest valid neighbors ----------------
__global__ __launch_bounds__(256) void knn_kernel(
    const float* __restrict__ pos, const unsigned char* __restrict__ mask,
    int* __restrict__ idx, int* __restrict__ nbv)
{
    int t = blockIdx.x * blockDim.x + threadIdx.x;
    if (t >= BN) return;
    int b = t / NN;
    float px = pos[t * 3 + 0], py = pos[t * 3 + 1], pz = pos[t * 3 + 2];
    const float* pb = pos + (size_t)b * NN * 3;
    const unsigned char* mb = mask + (size_t)b * NN;

    float bd[KK]; int bi[KK];
#pragma unroll
    for (int i = 0; i < KK; ++i) { bd[i] = __builtin_inff(); bi[i] = 0; }

    for (int j = 0; j < NN; ++j) {
        if (!mb[j]) continue;
        float dx = px - pb[j * 3 + 0];
        float dy = py - pb[j * 3 + 1];
        float dz = pz - pb[j * 3 + 2];
        float d = dx * dx + dy * dy + dz * dz;
        if (d < bd[KK - 1]) {
            bd[KK - 1] = d; bi[KK - 1] = j;
#pragma unroll
            for (int s = KK - 1; s > 0; --s) {
                if (bd[s] < bd[s - 1]) {
                    float td = bd[s]; bd[s] = bd[s - 1]; bd[s - 1] = td;
                    int   ti = bi[s]; bi[s] = bi[s - 1]; bi[s - 1] = ti;
                }
            }
        }
    }
#pragma unroll
    for (int i = 0; i < KK; ++i) {
        idx[(size_t)t * KK + i] = bi[i];
        nbv[(size_t)t * KK + i] = (bd[i] < 3.0e38f) ? 1 : 0;
    }
}

// ---------------- sine positional embedding (y-half then x-half) ------
__global__ __launch_bounds__(256) void sine_kernel(
    const float* __restrict__ pos, float* __restrict__ pe)
{
    size_t t = (size_t)blockIdx.x * blockDim.x + threadIdx.x;
    if (t >= (size_t)BN * DD) return;
    int e  = (int)(t % DD);
    int bn = (int)(t / DD);
    int coord = (e < DD / 2) ? 1 : 0;           // first half = y, second = x
    int w = e & (DD / 2 - 1);
    int j = w >> 1;
    float p = pos[(size_t)bn * 3 + coord];
    const float TWO_PI = 6.283185307179586f;
    const float L2_1E4 = 13.287712379549449f;   // log2(10000)
    float freq = TWO_PI * exp2f(-(float)j * (L2_1E4 / 64.0f));
    float th = p * freq;
    pe[t] = (w & 1) ? __cosf(th) : __sinf(th);
}

// ---------------- small conversion / elementwise kernels --------------
__global__ __launch_bounds__(256) void cvt_bf16_kernel(
    __bf16* __restrict__ dst, const float* __restrict__ src, int n)
{
    int t = blockIdx.x * blockDim.x + threadIdx.x;
    if (t < n) dst[t] = f2bf(src[t]);
}

__global__ __launch_bounds__(256) void add_cvt_bf16_kernel(
    __bf16* __restrict__ dst, const float* __restrict__ a,
    const float* __restrict__ b, int n)
{
    int t = blockIdx.x * blockDim.x + threadIdx.x;
    if (t < n) dst[t] = f2bf(a[t] + b[t]);
}

__global__ __launch_bounds__(256) void maskout_kernel(
    float* __restrict__ out, const float* __restrict__ h,
    const unsigned char* __restrict__ mask)
{
    size_t t = (size_t)blockIdx.x * blockDim.x + threadIdx.x;
    if (t >= (size_t)BN * DD) return;
    int bn = (int)(t / DD);
    out[t] = mask[bn] ? h[t] : 0.0f;
}

// ---------------- WMMA bf16 GEMM:  out = A[M,K] * W[N,K]^T ------------
// A lane fragment (16-bit A 16x32 layout): rows m, K chunks [kh*8..+7]
// and [16+kh*8..+7] -> two contiguous 16B global loads.
// B tile (weights) is streamed into LDS with CDNA5 async copies
// (GLOBAL_LOAD_ASYNC_TO_LDS_B128, ASYNCcnt) and double-buffered; each
// lane's B fragment is two 16B chunks [kh*8..+7] and [16+kh*8..+7] of an LDS tile row.
__device__ __forceinline__ bf16x16 load_frag_a(const __bf16* p, int kh) {
    bf16x8 lo = *(const bf16x8*)(p + kh * 8);
    bf16x8 hi = *(const bf16x8*)(p + 16 + kh * 8);
    return __builtin_shufflevector(lo, hi, 0,1,2,3,4,5,6,7,8,9,10,11,12,13,14,15);
}
__device__ __forceinline__ bf16x16 load_frag_lds(const __bf16* p) {
    bf16x8 lo = *(const bf16x8*)(p);
    bf16x8 hi = *(const bf16x8*)(p + 16);
    return __builtin_shufflevector(lo, hi, 0,1,2,3,4,5,6,7,8,9,10,11,12,13,14,15);
}
__device__ __forceinline__ void async_copy16(unsigned lds_off, const __bf16* g) {
    unsigned long long ga = (unsigned long long)(size_t)g;
    asm volatile("global_load_async_to_lds_b128 %0, %1, off"
                 :: "v"(lds_off), "v"(ga) : "memory");
}

__global__ __launch_bounds__(128) void gemm_bf16_kernel(
    const __bf16* __restrict__ A,   // [M][K] row-major
    const __bf16* __restrict__ W,   // [Nout][K] row-major (B = W^T)
    const float*  __restrict__ bias,// [Nout] or null
    float*  __restrict__ outF,      // [M][Nout] or null
    __bf16* __restrict__ outB,      // [M][Nout] or null
    int M, int Nout, int Kdim, float scale, int relu)
{
    __shared__ __align__(16) __bf16 bt[2][64][32];  // double-buffered 64x32 B tile

    int tid  = threadIdx.x;
    int lane = tid & 31;
    int wave = tid >> 5;
    int rowBlk = blockIdx.x * 64 + (wave >> 1) * 32;
    int colB0  = blockIdx.y * 64;
    int colBlk = colB0 + (wave & 1) * 32;
    int lo16 = lane & 15;
    int kh   = lane >> 4;

    const __bf16* Ar0 = A + (size_t)(rowBlk + lo16) * Kdim;
    const __bf16* Ar1 = A + (size_t)(rowBlk + 16 + lo16) * Kdim;

    // async-copy assignment: 64x32 bf16 tile = 256 x 16B chunks; 128 threads
    // copy 2 chunks each. chunk c: row = c>>2, quarter = c&3.
    int crow = tid >> 2, cq = tid & 3;
    const __bf16* g0 = W + (size_t)(colB0 + crow) * Kdim + cq * 8;
    const __bf16* g1 = g0 + (size_t)32 * Kdim;      // rows +32
    unsigned ldsBase = (unsigned)(size_t)(&bt[0][0][0]);
    unsigned l0 = ldsBase + (unsigned)(crow * 64 + cq * 16);
    unsigned l1 = l0 + 32u * 64u;
    const unsigned bufStride = 64u * 64u;           // 4 KB

    // prologue: stream tile 0 into buffer 0
    async_copy16(l0, g0);
    async_copy16(l1, g1);

    v8f acc00 = 0.0f, acc01 = 0.0f, acc10 = 0.0f, acc11 = 0.0f;

    int pb = 0;
    for (int kk = 0; kk < Kdim; kk += 32, pb ^= 1) {
        if (kk + 32 < Kdim) {
            // stream tile k+1 into the other buffer while computing tile k
            unsigned nb = (unsigned)(pb ^ 1) * bufStride;
            async_copy16(l0 + nb, g0 + kk + 32);
            async_copy16(l1 + nb, g1 + kk + 32);
            __builtin_prefetch((const void*)(Ar0 + kk + 32), 0, 1);
            __builtin_prefetch((const void*)(Ar1 + kk + 32), 0, 1);
            asm volatile("s_wait_asynccnt 0x2" ::: "memory");  // tile k done
        } else {
            asm volatile("s_wait_asynccnt 0x0" ::: "memory");
        }
        __syncthreads();    // all waves' async chunks of tile k visible

        bf16x16 a0 = load_frag_a(Ar0 + kk, kh);
        bf16x16 a1 = load_frag_a(Ar1 + kk, kh);
        bf16x16 b0 = load_frag_lds(&bt[pb][(wave & 1) * 32 + lo16][kh * 8]);
        bf16x16 b1 = load_frag_lds(&bt[pb][(wave & 1) * 32 + 16 + lo16][kh * 8]);

        acc00 = __builtin_amdgcn_wmma_f32_16x16x32_bf16(false, a0, false, b0, (short)0, acc00, false, false);
        acc01 = __builtin_amdgcn_wmma_f32_16x16x32_bf16(false, a0, false, b1, (short)0, acc01, false, false);
        acc10 = __builtin_amdgcn_wmma_f32_16x16x32_bf16(false, a1, false, b0, (short)0, acc10, false, false);
        acc11 = __builtin_amdgcn_wmma_f32_16x16x32_bf16(false, a1, false, b1, (short)0, acc11, false, false);

        __syncthreads();    // reads of tile k done before buffer is overwritten
    }

    // D layout: vgpr v -> m = v + 8*kh within tile, n = lane&15
    v8f accs[2][2] = { { acc00, acc01 }, { acc10, acc11 } };
#pragma unroll
    for (int mt = 0; mt < 2; ++mt) {
#pragma unroll
        for (int nt = 0; nt < 2; ++nt) {
            int col = colBlk + nt * 16 + lo16;
            float bv = bias ? bias[col] : 0.0f;
#pragma unroll
            for (int v = 0; v < 8; ++v) {
                int row = rowBlk + mt * 16 + v + 8 * kh;
                float val = (accs[mt][nt][v] + bv) * scale;
                if (relu) val = fmaxf(val, 0.0f);
                size_t o = (size_t)row * Nout + col;
                if (outF) outF[o] = val;
                if (outB) outB[o] = f2bf(val);
            }
        }
    }
}

// ---------------- local KNN attention (per (b,n,head) thread) ---------
__global__ __launch_bounds__(256) void attn_kernel(
    const float* __restrict__ q, const float* __restrict__ k,
    const float* __restrict__ v, const int* __restrict__ idx,
    const int* __restrict__ nbv, float* __restrict__ out)
{
    int t = blockIdx.x * blockDim.x + threadIdx.x;
    if (t >= BN * HH) return;
    int hh = t & (HH - 1);
    int bn = t >> 3;            // /HH
    int b  = bn >> 9;           // /NN

    const float4* qp = (const float4*)(q + (size_t)bn * DD + hh * HD);
    float4 qr[HD / 4];
#pragma unroll
    for (int i = 0; i < HD / 4; ++i) qr[i] = qp[i];

    const int* ip = idx + (size_t)bn * KK;
    const int* vp = nbv + (size_t)bn * KK;
    size_t base = (size_t)b * NN * DD + hh * HD;

    float sc[KK];
    float mx = -1e30f;
#pragma unroll
    for (int kk = 0; kk < KK; ++kk) {
        int j = ip[kk];
        const float4* kp = (const float4*)(k + base + (size_t)j * DD);
        float s = 0.0f;
#pragma unroll
        for (int i = 0; i < HD / 4; ++i) {
            float4 kv = kp[i];
            s += qr[i].x * kv.x + qr[i].y * kv.y + qr[i].z * kv.z + qr[i].w * kv.w;
        }
        s = vp[kk] ? s : -1e9f;
        sc[kk] = s;
        mx = fmaxf(mx, s);
    }
    float den = 0.0f;
#pragma unroll
    for (int kk = 0; kk < KK; ++kk) { sc[kk] = __expf(sc[kk] - mx); den += sc[kk]; }
    float inv = 1.0f / den;

    float4 o[HD / 4];
#pragma unroll
    for (int i = 0; i < HD / 4; ++i) { o[i].x = 0; o[i].y = 0; o[i].z = 0; o[i].w = 0; }
#pragma unroll
    for (int kk = 0; kk < KK; ++kk) {
        float a = sc[kk] * inv;
        int j = ip[kk];
        const float4* vr = (const float4*)(v + base + (size_t)j * DD);
#pragma unroll
        for (int i = 0; i < HD / 4; ++i) {
            float4 vv = vr[i];
            o[i].x += a * vv.x; o[i].y += a * vv.y; o[i].z += a * vv.z; o[i].w += a * vv.w;
        }
    }
    float4* op = (float4*)(out + (size_t)bn * DD + hh * HD);
#pragma unroll
    for (int i = 0; i < HD / 4; ++i) op[i] = o[i];
}

// ---------------- residual + LayerNorm (one wave per row) -------------
__global__ __launch_bounds__(128) void resid_ln_kernel(
    float* __restrict__ h, const float* __restrict__ delta,
    const float* __restrict__ gam, const float* __restrict__ bet)
{
    int lane = threadIdx.x & 31;
    int row  = blockIdx.x * (blockDim.x >> 5) + (threadIdx.x >> 5);
    if (row >= BN) return;
    const int PER = DD / 32;    // 8
    size_t base = (size_t)row * DD + lane * PER;

    float vals[PER];
    float sum = 0.0f;
#pragma unroll
    for (int i = 0; i < PER; ++i) { vals[i] = h[base + i] + delta[base + i]; sum += vals[i]; }
#pragma unroll
    for (int m = 16; m > 0; m >>= 1) sum += __shfl_xor(sum, m, 32);
    float mu = sum * (1.0f / DD);

    float vs = 0.0f;
#pragma unroll
    for (int i = 0; i < PER; ++i) { float d = vals[i] - mu; vs += d * d; }
#pragma unroll
    for (int m = 16; m > 0; m >>= 1) vs += __shfl_xor(vs, m, 32);
    float inv = rsqrtf(vs * (1.0f / DD) + 1e-5f);

#pragma unroll
    for (int i = 0; i < PER; ++i) {
        int c = lane * PER + i;
        h[base + i] = (vals[i] - mu) * inv * gam[c] + bet[c];
    }
}

// ======================================================================
extern "C" void kernel_launch(void* const* d_in, const int* in_sizes, int n_in,
                              void* d_out, int out_size, void* d_ws, size_t ws_size,
                              hipStream_t stream) {
    (void)in_sizes; (void)n_in; (void)out_size; (void)ws_size;
    const float*         x     = (const float*)d_in[0];
    const unsigned char* xmask = (const unsigned char*)d_in[1];
    const float*         xpos  = (const float*)d_in[2];
    const float* Wq = (const float*)d_in[3];  const float* bq = (const float*)d_in[4];
    const float* Wk = (const float*)d_in[5];  const float* bk = (const float*)d_in[6];
    const float* Wv = (const float*)d_in[7];  const float* bv = (const float*)d_in[8];
    const float* Wo = (const float*)d_in[9];  const float* bo = (const float*)d_in[10];
    const float* W1 = (const float*)d_in[11]; const float* b1 = (const float*)d_in[12];
    const float* W2 = (const float*)d_in[13]; const float* b2 = (const float*)d_in[14];
    const float* ln1s = (const float*)d_in[15]; const float* ln1b = (const float*)d_in[16];
    const float* ln2s = (const float*)d_in[17]; const float* ln2b = (const float*)d_in[18];
    float* out = (float*)d_out;

    char* ws = (char*)d_ws;
    size_t off = 0;
#define WS_ALLOC(ty, name, bytes) ty name = (ty)(ws + off); off += (((size_t)(bytes)) + 255) & ~(size_t)255
    WS_ALLOC(float*,  pe,    (size_t)BN * DD * 4);
    WS_ALLOC(int*,    idx,   (size_t)BN * KK * 4);
    WS_ALLOC(int*,    nbv,   (size_t)BN * KK * 4);
    WS_ALLOC(float*,  hbuf,  (size_t)BN * DD * 4);
    WS_ALLOC(__bf16*, abuf,  (size_t)BN * DD * 2);   // qk / h (FF input) in bf16
    WS_ALLOC(__bf16*, hbb,   (size_t)BN * DD * 2);   // h / attn-out in bf16
    WS_ALLOC(float*,  qbuf,  (size_t)BN * DD * 4);
    WS_ALLOC(float*,  kbuf,  (size_t)BN * DD * 4);   // reused as proj-out
    WS_ALLOC(float*,  vbuf,  (size_t)BN * DD * 4);   // reused as ff2-out
    WS_ALLOC(float*,  att,   (size_t)BN * DD * 4);
    WS_ALLOC(__bf16*, ffb,   (size_t)BN * DFF * 2);  // relu(FF1) in bf16
    WS_ALLOC(__bf16*, wbuf,  (size_t)(4 * DD * DD + 2 * DD * DFF) * 2);
#undef WS_ALLOC
    __bf16* wq = wbuf;
    __bf16* wk = wq + (size_t)DD * DD;
    __bf16* wv = wk + (size_t)DD * DD;
    __bf16* wo = wv + (size_t)DD * DD;
    __bf16* w1 = wo + (size_t)DD * DD;
    __bf16* w2 = w1 + (size_t)DFF * DD;

    const float scale = 1.0f / sqrtf((float)HD);
    const int ND = BN * DD;

    // h = x
    hipMemcpyAsync(hbuf, x, (size_t)ND * 4, hipMemcpyDeviceToDevice, stream);
    // KNN + positional embedding (layer-invariant)
    knn_kernel<<<BN / 256, 256, 0, stream>>>(xpos, xmask, idx, nbv);
    sine_kernel<<<ND / 256, 256, 0, stream>>>(xpos, pe);

    dim3 gblk(128);
    dim3 gD(BN / 64, DD / 64);
    dim3 gF(BN / 64, DFF / 64);

    for (int l = 0; l < LL; ++l) {
        // weights -> bf16
        cvt_bf16_kernel<<<(DD * DD) / 256, 256, 0, stream>>>(wq, Wq + (size_t)l * DD * DD, DD * DD);
        cvt_bf16_kernel<<<(DD * DD) / 256, 256, 0, stream>>>(wk, Wk + (size_t)l * DD * DD, DD * DD);
        cvt_bf16_kernel<<<(DD * DD) / 256, 256, 0, stream>>>(wv, Wv + (size_t)l * DD * DD, DD * DD);
        cvt_bf16_kernel<<<(DD * DD) / 256, 256, 0, stream>>>(wo, Wo + (size_t)l * DD * DD, DD * DD);
        cvt_bf16_kernel<<<(DFF * DD) / 256, 256, 0, stream>>>(w1, W1 + (size_t)l * DFF * DD, DFF * DD);
        cvt_bf16_kernel<<<(DD * DFF) / 256, 256, 0, stream>>>(w2, W2 + (size_t)l * DD * DFF, DD * DFF);

        // qk = bf16(h + pos_emb); hbb = bf16(h)
        add_cvt_bf16_kernel<<<ND / 256, 256, 0, stream>>>(abuf, hbuf, pe, ND);
        cvt_bf16_kernel<<<ND / 256, 256, 0, stream>>>(hbb, hbuf, ND);

        // projections (WMMA GEMMs, async-LDS weight streaming)
        gemm_bf16_kernel<<<gD, gblk, 0, stream>>>(abuf, wq, bq + (size_t)l * DD, qbuf, nullptr, BN, DD, DD, scale, 0);
        gemm_bf16_kernel<<<gD, gblk, 0, stream>>>(abuf, wk, bk + (size_t)l * DD, kbuf, nullptr, BN, DD, DD, 1.0f, 0);
        gemm_bf16_kernel<<<gD, gblk, 0, stream>>>(hbb,  wv, bv + (size_t)l * DD, vbuf, nullptr, BN, DD, DD, 1.0f, 0);

        // neighborhood attention
        attn_kernel<<<(BN * HH) / 256, 256, 0, stream>>>(qbuf, kbuf, vbuf, idx, nbv, att);

        // output projection + residual LN1
        cvt_bf16_kernel<<<ND / 256, 256, 0, stream>>>(hbb, att, ND);
        gemm_bf16_kernel<<<gD, gblk, 0, stream>>>(hbb, wo, bo + (size_t)l * DD, kbuf, nullptr, BN, DD, DD, 1.0f, 0);
        resid_ln_kernel<<<BN / 4, 128, 0, stream>>>(hbuf, kbuf, ln1s + (size_t)l * DD, ln1b + (size_t)l * DD);

        // FFN: relu(h@W1^T+b1)@W2^T+b2, residual LN2
        cvt_bf16_kernel<<<ND / 256, 256, 0, stream>>>(abuf, hbuf, ND);
        gemm_bf16_kernel<<<gF, gblk, 0, stream>>>(abuf, w1, b1 + (size_t)l * DFF, nullptr, ffb, BN, DFF, DD, 1.0f, 1);
        gemm_bf16_kernel<<<gD, gblk, 0, stream>>>(ffb, w2, b2 + (size_t)l * DD, vbuf, nullptr, BN, DD, DFF, 1.0f, 0);
        resid_ln_kernel<<<BN / 4, 128, 0, stream>>>(hbuf, vbuf, ln2s + (size_t)l * DD, ln2b + (size_t)l * DD);
    }

    // zero out invalid rows
    maskout_kernel<<<ND / 256, 256, 0, stream>>>(out, hbuf, xmask);
}
